// MERIT_79602923864615
// MI455X (gfx1250) — matrix operand order repaired and checked by reading, not compiled, not run on hardware
//
#include <hip/hip_runtime.h>
#include <hip/hip_bf16.h>

// ---------------------------------------------------------------------------
// Types for CDNA5 WMMA (wave32, v_wmma_f32_16x16x32_bf16)
// ---------------------------------------------------------------------------
typedef __attribute__((ext_vector_type(16))) __bf16 v16bf;
typedef __attribute__((ext_vector_type(2)))  __bf16 v2bf;
typedef __attribute__((ext_vector_type(8)))  float  v8f;

union FragBF {
    v16bf v;
    uint4 q[2];     // two 16B spans -> ds_load_b128 pair
};

// fp32 -> bf16 via the compiler's native conversion (hardware cvt on gfx1250)
__device__ __forceinline__ unsigned short f2bf(float f) {
    union { __bf16 b; unsigned short u; } cv;
    cv.b = (__bf16)f;
    return cv.u;
}

// pack two fp32 -> one dword of two bf16
__device__ __forceinline__ unsigned int pack2bf(float lo, float hi) {
#if __has_builtin(__builtin_amdgcn_cvt_pk_bf16_f32)
    v2bf p = __builtin_amdgcn_cvt_pk_bf16_f32(lo, hi);
    union { v2bf v; unsigned int u; } cv;
    cv.v = p;
    return cv.u;
#else
    union { v2bf v; unsigned int u; } cv;
    cv.v[0] = (__bf16)lo;
    cv.v[1] = (__bf16)hi;
    return cv.u;
#endif
}

// ---------------------------------------------------------------------------
// GEMM: C[M,N] = A[M,K] @ B[K,N] (+ bias[N])   fp32 in/out, bf16 WMMA compute
// Block tile 128x128, K-tile 32, 256 threads = 8 waves (2x4), wave tile 64x32.
// LDS row stride 40 halves = 80 B: rows 16B-aligned (b128 frag loads) and
// 20-bank stride -> conflict-free across the 16 lanes of a fragment load.
// Register-level software pipeline: next K-tile prefetched into VGPRs while
// WMMAs consume the current LDS tile.
// ---------------------------------------------------------------------------
#define TILE_M 128
#define TILE_N 128
#define TILE_K 32
#define LDSS   40

__global__ __launch_bounds__(256)
void gemm_bf16wmma_kernel(const float* __restrict__ A, const float* __restrict__ B,
                          const float* bias, float* __restrict__ C,
                          int M, int N, int K)
{
    __shared__ unsigned short sA[TILE_M * LDSS];
    __shared__ unsigned short sB[TILE_N * LDSS];   // stores B^T tile: [n][k]

    const int t     = threadIdx.x;
    const int lane  = t & 31;
    const int wave  = t >> 5;
    const int grp   = lane >> 4;         // half-wave group
    const int mr    = lane & 15;
    const int wmB   = (wave >> 2) * 64;  // wave M base (0 or 64)
    const int wnB   = (wave & 3) * 32;   // wave N base (0,32,64,96)

    const int m0 = blockIdx.y * TILE_M;
    const int n0 = blockIdx.x * TILE_N;

    v8f acc[4][2];
#pragma unroll
    for (int i = 0; i < 4; ++i)
#pragma unroll
        for (int j = 0; j < 2; ++j)
#pragma unroll
            for (int r = 0; r < 8; ++r) acc[i][j][r] = 0.0f;

    const int aRow = t >> 3;          // 0..31
    const int aCol = (t & 7) * 4;     // 0..28
    const int bK   = t >> 5;          // 0..7
    const int bN   = (t & 31) * 4;    // 0..124

    // prefetch K-tile 0 into registers
    float4 ra[4], rb[4];
#pragma unroll
    for (int it = 0; it < 4; ++it) {
        ra[it] = *reinterpret_cast<const float4*>(
            A + (size_t)(m0 + it * 32 + aRow) * K + aCol);
        rb[it] = *reinterpret_cast<const float4*>(
            B + (size_t)(it * 8 + bK) * N + n0 + bN);
    }

    for (int k0 = 0; k0 < K; k0 += TILE_K) {
        // stage registers -> LDS (A as packed b64 stores, B transposed)
#pragma unroll
        for (int it = 0; it < 4; ++it) {
            int row = it * 32 + aRow;
            uint2 p;
            p.x = pack2bf(ra[it].x, ra[it].y);
            p.y = pack2bf(ra[it].z, ra[it].w);
            *reinterpret_cast<uint2*>(&sA[row * LDSS + aCol]) = p;
            int k = it * 8 + bK;
            sB[(bN + 0) * LDSS + k] = f2bf(rb[it].x);
            sB[(bN + 1) * LDSS + k] = f2bf(rb[it].y);
            sB[(bN + 2) * LDSS + k] = f2bf(rb[it].z);
            sB[(bN + 3) * LDSS + k] = f2bf(rb[it].w);
        }
        __syncthreads();

        // prefetch next K-tile while WMMAs run on the current one
        if (k0 + TILE_K < K) {
            int k1 = k0 + TILE_K;
#pragma unroll
            for (int it = 0; it < 4; ++it) {
                ra[it] = *reinterpret_cast<const float4*>(
                    A + (size_t)(m0 + it * 32 + aRow) * K + k1 + aCol);
                rb[it] = *reinterpret_cast<const float4*>(
                    B + (size_t)(k1 + it * 8 + bK) * N + n0 + bN);
            }
        }

        // fragments: each = two contiguous 16B LDS spans (ds_load_b128)
        FragBF fa[4], fb[2];
#pragma unroll
        for (int i = 0; i < 4; ++i) {
            const unsigned short* base = sA + (wmB + 16 * i + mr) * LDSS;
            fa[i].q[0] = *reinterpret_cast<const uint4*>(base + 8 * grp);
            fa[i].q[1] = *reinterpret_cast<const uint4*>(base + 16 + 8 * grp);
        }
#pragma unroll
        for (int j = 0; j < 2; ++j) {
            const unsigned short* base = sB + (wnB + 16 * j + mr) * LDSS;
            fb[j].q[0] = *reinterpret_cast<const uint4*>(base + 8 * grp);
            fb[j].q[1] = *reinterpret_cast<const uint4*>(base + 16 + 8 * grp);
        }
#pragma unroll
        for (int i = 0; i < 4; ++i)
#pragma unroll
            for (int j = 0; j < 2; ++j)
                acc[i][j] = __builtin_amdgcn_wmma_f32_16x16x32_bf16(
                    false, fa[i].v, false, fb[j].v, (short)0, acc[i][j], false, false);
        __syncthreads();
    }

    // epilogue: C layout -> lane (grp,mr), VGPR r holds (M = r + 8*grp, N = mr)
#pragma unroll
    for (int i = 0; i < 4; ++i) {
#pragma unroll
        for (int j = 0; j < 2; ++j) {
            int n = n0 + wnB + 16 * j + mr;
            float bv = bias ? bias[n] : 0.0f;
#pragma unroll
            for (int r = 0; r < 8; ++r) {
                int m = m0 + wmB + 16 * i + r + 8 * grp;
                C[(size_t)m * N + n] = acc[i][j][r] + bv;
            }
        }
    }
}

// ---------------------------------------------------------------------------
// Gram/phi kernel: for unit-row matrix HN [Nrows x D] compute
//   phi_raw[m] += sum_n exp( HN[m,:] . HN[n,:] )  over this block's n-tile.
// Same WMMA tiling; exp + row reduction fused, Gram never materialized.
// ---------------------------------------------------------------------------
__global__ __launch_bounds__(256)
void gram_phi_kernel(const float* __restrict__ HN, float* __restrict__ phi,
                     int Nrows, int D)
{
    __shared__ unsigned short sA[TILE_M * LDSS];
    __shared__ unsigned short sB[TILE_N * LDSS];

    const int t    = threadIdx.x;
    const int lane = t & 31;
    const int wave = t >> 5;
    const int grp  = lane >> 4;
    const int mr   = lane & 15;
    const int wmB  = (wave >> 2) * 64;
    const int wnB  = (wave & 3) * 32;

    const int m0 = blockIdx.y * TILE_M;
    const int n0 = blockIdx.x * TILE_N;

    v8f acc[4][2];
#pragma unroll
    for (int i = 0; i < 4; ++i)
#pragma unroll
        for (int j = 0; j < 2; ++j)
#pragma unroll
            for (int r = 0; r < 8; ++r) acc[i][j][r] = 0.0f;

    const int aRow = t >> 3;
    const int aCol = (t & 7) * 4;

    // prefetch K-tile 0 (both operands are rows of HN: B = HN^T)
    float4 rm[4], rn[4];
#pragma unroll
    for (int it = 0; it < 4; ++it) {
        rm[it] = *reinterpret_cast<const float4*>(
            HN + (size_t)(m0 + it * 32 + aRow) * D + aCol);
        rn[it] = *reinterpret_cast<const float4*>(
            HN + (size_t)(n0 + it * 32 + aRow) * D + aCol);
    }

    for (int k0 = 0; k0 < D; k0 += TILE_K) {
#pragma unroll
        for (int it = 0; it < 4; ++it) {
            int row = it * 32 + aRow;
            uint2 pm, pn;
            pm.x = pack2bf(rm[it].x, rm[it].y);
            pm.y = pack2bf(rm[it].z, rm[it].w);
            pn.x = pack2bf(rn[it].x, rn[it].y);
            pn.y = pack2bf(rn[it].z, rn[it].w);
            *reinterpret_cast<uint2*>(&sA[row * LDSS + aCol]) = pm;
            *reinterpret_cast<uint2*>(&sB[row * LDSS + aCol]) = pn;
        }
        __syncthreads();

        if (k0 + TILE_K < D) {
            int k1 = k0 + TILE_K;
#pragma unroll
            for (int it = 0; it < 4; ++it) {
                rm[it] = *reinterpret_cast<const float4*>(
                    HN + (size_t)(m0 + it * 32 + aRow) * D + k1 + aCol);
                rn[it] = *reinterpret_cast<const float4*>(
                    HN + (size_t)(n0 + it * 32 + aRow) * D + k1 + aCol);
            }
        }

        FragBF fa[4], fb[2];
#pragma unroll
        for (int i = 0; i < 4; ++i) {
            const unsigned short* base = sA + (wmB + 16 * i + mr) * LDSS;
            fa[i].q[0] = *reinterpret_cast<const uint4*>(base + 8 * grp);
            fa[i].q[1] = *reinterpret_cast<const uint4*>(base + 16 + 8 * grp);
        }
#pragma unroll
        for (int j = 0; j < 2; ++j) {
            const unsigned short* base = sB + (wnB + 16 * j + mr) * LDSS;
            fb[j].q[0] = *reinterpret_cast<const uint4*>(base + 8 * grp);
            fb[j].q[1] = *reinterpret_cast<const uint4*>(base + 16 + 8 * grp);
        }
#pragma unroll
        for (int i = 0; i < 4; ++i)
#pragma unroll
            for (int j = 0; j < 2; ++j)
                acc[i][j] = __builtin_amdgcn_wmma_f32_16x16x32_bf16(
                    false, fa[i].v, false, fb[j].v, (short)0, acc[i][j], false, false);
        __syncthreads();
    }

    // exp + row-sum: lanes within each 16-lane half hold N=0..15 for row r+8*grp
#pragma unroll
    for (int i = 0; i < 4; ++i) {
#pragma unroll
        for (int r = 0; r < 8; ++r) {
            float s = expf(acc[i][0][r]) + expf(acc[i][1][r]);
#pragma unroll
            for (int off = 1; off < 16; off <<= 1) s += __shfl_xor(s, off, 32);
            if (mr == 0)
                atomicAdd(&phi[m0 + wmB + 16 * i + 8 * grp + r], s);
        }
    }
}

// ---------------------------------------------------------------------------
// BatchNorm batch stats: per-column sum / sumsq with atomics
// ---------------------------------------------------------------------------
__global__ __launch_bounds__(256)
void bn_stats_kernel(const float* __restrict__ H, float* __restrict__ sum,
                     float* __restrict__ sumsq, int rows, int cols, int rowsPerBlock)
{
    int c  = blockIdx.x * blockDim.x + threadIdx.x;
    int r0 = blockIdx.y * rowsPerBlock;
    float s = 0.0f, ss = 0.0f;
    for (int r = r0; r < r0 + rowsPerBlock; ++r) {
        float v = H[(size_t)r * cols + c];
        s += v; ss += v * v;
    }
    atomicAdd(&sum[c], s);
    atomicAdd(&sumsq[c], ss);
}

// BN normalize (training-mode batch stats) + PReLU, in place
__global__ __launch_bounds__(256)
void bn_apply_kernel(float* __restrict__ H, const float* __restrict__ sum,
                     const float* __restrict__ sumsq, const float* __restrict__ g,
                     const float* __restrict__ be, const float* __restrict__ a,
                     int rows, int cols)
{
    size_t idx = (size_t)blockIdx.x * blockDim.x + threadIdx.x;
    size_t total = (size_t)rows * cols;
    if (idx >= total) return;
    int c = (int)(idx % cols);
    float invN = 1.0f / (float)rows;
    float mu  = sum[c] * invN;
    float var = sumsq[c] * invN - mu * mu;
    float y = (H[idx] - mu) * rsqrtf(var + 1e-5f) * g[c] + be[c];
    float slope = a[0];
    H[idx] = (y >= 0.0f) ? y : slope * y;
}

// ---------------------------------------------------------------------------
// Row-wise cosine similarity: out[r] = a_r.b_r / max(|a_r||b_r|, eps)
// ---------------------------------------------------------------------------
__global__ __launch_bounds__(128)
void rowsim_kernel(const float* __restrict__ a, const float* __restrict__ b,
                   float* __restrict__ out, int D)
{
    int r = blockIdx.x;
    const float* ar = a + (size_t)r * D;
    const float* br = b + (size_t)r * D;
    float d = 0.0f, na = 0.0f, nb = 0.0f;
    for (int c = threadIdx.x; c < D; c += 128) {
        float x = ar[c], y = br[c];
        d += x * y; na += x * x; nb += y * y;
    }
#pragma unroll
    for (int off = 16; off > 0; off >>= 1) {
        d  += __shfl_xor(d, off, 32);
        na += __shfl_xor(na, off, 32);
        nb += __shfl_xor(nb, off, 32);
    }
    __shared__ float sd[4], sa2[4], sb2[4];
    int wid = threadIdx.x >> 5, lane = threadIdx.x & 31;
    if (lane == 0) { sd[wid] = d; sa2[wid] = na; sb2[wid] = nb; }
    __syncthreads();
    if (threadIdx.x == 0) {
        float D0 = sd[0] + sd[1] + sd[2] + sd[3];
        float A0 = sa2[0] + sa2[1] + sa2[2] + sa2[3];
        float B0 = sb2[0] + sb2[1] + sb2[2] + sb2[3];
        out[r] = D0 / fmaxf(sqrtf(A0) * sqrtf(B0), 1e-8f);
    }
}

// Row normalization: hn_r = a_r / max(|a_r|, eps)
__global__ __launch_bounds__(128)
void rownorm_kernel(const float* __restrict__ a, float* __restrict__ hn, int D)
{
    int r = blockIdx.x;
    const float* ar = a + (size_t)r * D;
    float na = 0.0f;
    for (int c = threadIdx.x; c < D; c += 128) { float x = ar[c]; na += x * x; }
#pragma unroll
    for (int off = 16; off > 0; off >>= 1) na += __shfl_xor(na, off, 32);
    __shared__ float sh[4];
    int wid = threadIdx.x >> 5, lane = threadIdx.x & 31;
    if (lane == 0) sh[wid] = na;
    __syncthreads();
    float inv = 1.0f / fmaxf(sqrtf(sh[0] + sh[1] + sh[2] + sh[3]), 1e-8f);
    float* hr = hn + (size_t)r * D;
    for (int c = threadIdx.x; c < D; c += 128) hr[c] = ar[c] * inv;
}

__global__ void zero_kernel(float* p, int n)
{
    int i = blockIdx.x * blockDim.x + threadIdx.x;
    if (i < n) p[i] = 0.0f;
}

// ---------------------------------------------------------------------------
// Final scalar reduction (single block)
// ---------------------------------------------------------------------------
__device__ __forceinline__ float blockReduceSum(float v, float* sh)
{
#pragma unroll
    for (int off = 16; off > 0; off >>= 1) v += __shfl_xor(v, off, 32);
    int wid = threadIdx.x >> 5, lane = threadIdx.x & 31;
    int nw = (blockDim.x + 31) >> 5;
    if (lane == 0) sh[wid] = v;
    __syncthreads();
    float r = (threadIdx.x < (unsigned)nw) ? sh[threadIdx.x] : 0.0f;
    if (threadIdx.x < 32) {
#pragma unroll
        for (int off = 16; off > 0; off >>= 1) r += __shfl_xor(r, off, 32);
        if (threadIdx.x == 0) sh[0] = r;
    }
    __syncthreads();
    float out = sh[0];
    __syncthreads();
    return out;
}

__device__ __forceinline__ float blockReduceMax(float v, float* sh)
{
#pragma unroll
    for (int off = 16; off > 0; off >>= 1) v = fmaxf(v, __shfl_xor(v, off, 32));
    int wid = threadIdx.x >> 5, lane = threadIdx.x & 31;
    int nw = (blockDim.x + 31) >> 5;
    if (lane == 0) sh[wid] = v;
    __syncthreads();
    float r = (threadIdx.x < (unsigned)nw) ? sh[threadIdx.x] : -3.0e38f;
    if (threadIdx.x < 32) {
#pragma unroll
        for (int off = 16; off > 0; off >>= 1) r = fmaxf(r, __shfl_xor(r, off, 32));
        if (threadIdx.x == 0) sh[0] = r;
    }
    __syncthreads();
    float out = sh[0];
    __syncthreads();
    return out;
}

__global__ __launch_bounds__(1024)
void final_loss_kernel(const float* __restrict__ s, const float* __restrict__ s12,
                       const float* __restrict__ s21, const float* __restrict__ phi1,
                       const float* __restrict__ phi2, float* __restrict__ out, int n)
{
    __shared__ float sh[32];
    const float E1 = 2.718281828459045f;   // diagonal of exp Gram (unit rows)
    int tid = threadIdx.x;

    float m_s = -3.0e38f, m_12 = -3.0e38f, m_21 = -3.0e38f;
    for (int i = tid; i < n; i += blockDim.x) {
        m_s  = fmaxf(m_s,  s[i]);
        m_12 = fmaxf(m_12, s12[i]);
        m_21 = fmaxf(m_21, s21[i]);
    }
    m_s  = blockReduceMax(m_s, sh);
    m_12 = blockReduceMax(m_12, sh);
    m_21 = blockReduceMax(m_21, sh);

    float se_s = 0, se_12 = 0, se_21 = 0;
    float sum_s = 0, sum_12 = 0, sum_21 = 0;
    float li1 = 0, li2 = 0;
    for (int i = tid; i < n; i += blockDim.x) {
        float v = s[i];
        se_s  += expf(v - m_s);
        sum_s += v;
        li1 += logf(expf(v) + (phi1[i] - E1)) - v;
        li2 += logf(expf(v) + (phi2[i] - E1)) - v;
        float v12 = s12[i]; se_12 += expf(v12 - m_12); sum_12 += v12;
        float v21 = s21[i]; se_21 += expf(v21 - m_21); sum_21 += v21;
    }
    se_s   = blockReduceSum(se_s, sh);
    sum_s  = blockReduceSum(sum_s, sh);
    li1    = blockReduceSum(li1, sh);
    li2    = blockReduceSum(li2, sh);
    se_12  = blockReduceSum(se_12, sh);
    sum_12 = blockReduceSum(sum_12, sh);
    se_21  = blockReduceSum(se_21, sh);
    sum_21 = blockReduceSum(sum_21, sh);

    if (tid == 0) {
        float fn = (float)n;
        float lse_s  = m_s  + logf(se_s);
        float lse12  = m_12 + logf(se_12);
        float lse21  = m_21 + logf(se_21);
        float sum_linter = fn * lse_s - sum_s;        // Σ -log_softmax(s)
        float lcv = sum_linter + 0.5f * (li1 + li2);
        float lcn = 0.5f * ((fn * lse12 - sum_12) + (fn * lse21 - sum_21));
        out[0] = 0.6f * lcv + 0.4f * lcn;
    }
}

// ---------------------------------------------------------------------------
// Host-side orchestration
// ---------------------------------------------------------------------------
static const int NROW = 8192, DD = 512, HH = 4096, PP = 512;

static void gemm(const float* A, const float* B, const float* bias, float* C,
                 int M, int N, int K, hipStream_t st)
{
    dim3 grid(N / TILE_N, M / TILE_M), block(256);
    gemm_bf16wmma_kernel<<<grid, block, 0, st>>>(A, B, bias, C, M, N, K);
}

static void zero_buf(float* p, int n, hipStream_t st)
{
    zero_kernel<<<(n + 255) / 256, 256, 0, st>>>(p, n);
}

// Linear -> BN(batch stats) -> PReLU -> Linear
static void run_mlp(const float* x, int K1,
                    const float* W1, const float* b1, const float* g,
                    const float* be, const float* a,
                    const float* W2, const float* b2,
                    float* Hbuf, float* bnsum, float* bnss, float* out,
                    hipStream_t st)
{
    gemm(x, W1, b1, Hbuf, NROW, HH, K1, st);
    zero_buf(bnsum, HH, st);
    zero_buf(bnss, HH, st);
    bn_stats_kernel<<<dim3(HH / 256, 16), 256, 0, st>>>(Hbuf, bnsum, bnss, NROW, HH, NROW / 16);
    size_t total = (size_t)NROW * HH;
    bn_apply_kernel<<<(unsigned)((total + 255) / 256), 256, 0, st>>>(
        Hbuf, bnsum, bnss, g, be, a, NROW, HH);
    gemm(Hbuf, W2, b2, out, NROW, PP, HH, st);
}

static void run_encode(const float* adj, const float* feat,
                       const float* Wg, const float* bg,
                       const float* W1, const float* b1, const float* g,
                       const float* be, const float* a,
                       const float* W2, const float* b2,
                       float* X, float* rep, float* Hbuf,
                       float* bnsum, float* bnss, float* out, hipStream_t st)
{
    gemm(feat, Wg, bg, X, NROW, DD, DD, st);          // feats @ Wg + bg
    gemm(adj, X, nullptr, rep, NROW, DD, NROW, st);   // adj @ X   (K = 8192)
    run_mlp(rep, DD, W1, b1, g, be, a, W2, b2, Hbuf, bnsum, bnss, out, st);
}

extern "C" void kernel_launch(void* const* d_in, const int* in_sizes, int n_in,
                              void* d_out, int out_size, void* d_ws, size_t ws_size,
                              hipStream_t stream)
{
    const float* adj1  = (const float*)d_in[0];
    const float* adj2  = (const float*)d_in[1];
    const float* feat1 = (const float*)d_in[2];
    const float* feat2 = (const float*)d_in[3];
    // online encoder params (4..12): Wg,bg,W1,b1,g1,be1,a1,W2,b2
    const float* oWg  = (const float*)d_in[4];
    const float* obg  = (const float*)d_in[5];
    const float* oW1  = (const float*)d_in[6];
    const float* ob1  = (const float*)d_in[7];
    const float* og1  = (const float*)d_in[8];
    const float* obe1 = (const float*)d_in[9];
    const float* oa1  = (const float*)d_in[10];
    const float* oW2  = (const float*)d_in[11];
    const float* ob2  = (const float*)d_in[12];
    // target encoder params (13..21)
    const float* tWg  = (const float*)d_in[13];
    const float* tbg  = (const float*)d_in[14];
    const float* tW1  = (const float*)d_in[15];
    const float* tb1  = (const float*)d_in[16];
    const float* tg1  = (const float*)d_in[17];
    const float* tbe1 = (const float*)d_in[18];
    const float* ta1  = (const float*)d_in[19];
    const float* tW2  = (const float*)d_in[20];
    const float* tb2  = (const float*)d_in[21];
    // projector params (22..28): pW1,pb1,pg1,pbe1,pa1,pW2,pb2
    const float* pW1  = (const float*)d_in[22];
    const float* pb1  = (const float*)d_in[23];
    const float* pg1  = (const float*)d_in[24];
    const float* pbe1 = (const float*)d_in[25];
    const float* pa1  = (const float*)d_in[26];
    const float* pW2  = (const float*)d_in[27];
    const float* pb2  = (const float*)d_in[28];
    (void)n_in; (void)in_sizes; (void)out_size; (void)ws_size;

    // ---- workspace carve (floats) ----
    float* ws = (float*)d_ws;
    size_t off = 0;
    auto carve = [&](size_t n) { float* p = ws + off; off += n; return p; };
    const size_t MAT = (size_t)NROW * PP;         // 8192*512
    float* X    = carve(MAT);
    float* rep  = carve(MAT);
    float* Hbuf = carve((size_t)NROW * HH);       // 8192*4096
    float* op1  = carve(MAT);
    float* op2  = carve(MAT);
    float* tp1  = carve(MAT);
    float* tp2  = carve(MAT);
    float* pr1  = carve(MAT);
    float* pr2  = carve(MAT);
    float* bnsum = carve(HH);
    float* bnss  = carve(HH);
    float* sv   = carve(NROW);
    float* s12  = carve(NROW);
    float* s21  = carve(NROW);
    float* phi1 = carve(NROW);
    float* phi2 = carve(NROW);
    float* hn1  = X;    // X / rep are dead after the encodes; recycle
    float* hn2  = rep;

    // ---- four encodes ----
    run_encode(adj1, feat1, oWg, obg, oW1, ob1, og1, obe1, oa1, oW2, ob2,
               X, rep, Hbuf, bnsum, bnss, op1, stream);
    run_encode(adj2, feat2, oWg, obg, oW1, ob1, og1, obe1, oa1, oW2, ob2,
               X, rep, Hbuf, bnsum, bnss, op2, stream);
    run_encode(adj1, feat1, tWg, tbg, tW1, tb1, tg1, tbe1, ta1, tW2, tb2,
               X, rep, Hbuf, bnsum, bnss, tp1, stream);
    run_encode(adj2, feat2, tWg, tbg, tW1, tb1, tg1, tbe1, ta1, tW2, tb2,
               X, rep, Hbuf, bnsum, bnss, tp2, stream);

    // ---- projector ----
    run_mlp(op1, PP, pW1, pb1, pg1, pbe1, pa1, pW2, pb2, Hbuf, bnsum, bnss, pr1, stream);
    run_mlp(op2, PP, pW1, pb1, pg1, pbe1, pa1, pW2, pb2, Hbuf, bnsum, bnss, pr2, stream);

    // ---- rowwise cosine sims ----
    rowsim_kernel<<<NROW, 128, 0, stream>>>(pr1, pr2, sv, PP);
    rowsim_kernel<<<NROW, 128, 0, stream>>>(pr1, tp2, s12, PP);
    rowsim_kernel<<<NROW, 128, 0, stream>>>(pr2, tp1, s21, PP);

    // ---- phi via fused exp-Gram row sums ----
    rownorm_kernel<<<NROW, 128, 0, stream>>>(pr1, hn1, PP);
    rownorm_kernel<<<NROW, 128, 0, stream>>>(pr2, hn2, PP);
    zero_buf(phi1, NROW, stream);
    zero_buf(phi2, NROW, stream);
    {
        dim3 grid(NROW / TILE_N, NROW / TILE_M), block(256);
        gram_phi_kernel<<<grid, block, 0, stream>>>(hn1, phi1, NROW, PP);
        gram_phi_kernel<<<grid, block, 0, stream>>>(hn2, phi2, NROW, PP);
    }

    // ---- final scalar ----
    final_loss_kernel<<<1, 1024, 0, stream>>>(sv, s12, s21, phi1, phi2,
                                              (float*)d_out, NROW);
}
